// EquivariantAttention_53231824666934
// MI455X (gfx1250) — matrix operand order repaired and checked by reading, not compiled, not run on hardware
//
#include <hip/hip_runtime.h>
#include <hip/hip_bf16.h>

// ---------- types ----------
typedef __attribute__((ext_vector_type(16))) __bf16        v16bf;
typedef __attribute__((ext_vector_type(8)))  float         v8f;
typedef __attribute__((ext_vector_type(4)))  unsigned int  u32x4;
typedef __attribute__((ext_vector_type(4)))  float         f32x4;

#define B_  16
#define C_  512
#define HW_ 4096     // 64*64
#define G_  4
#define CG_ 128      // C/G
#define QKVC 1536    // 3*C

// ---------- helpers ----------
__device__ __forceinline__ unsigned short f32_to_bf16(float f) {
    unsigned int u = __float_as_uint(f);
    unsigned int r = u + 0x7FFFu + ((u >> 16) & 1u);   // round-to-nearest-even
    return (unsigned short)(r >> 16);
}

// A-matrix (16x32 bf16, MxK) tile from row-major LDS [row=M][col=K], stride ld (elems).
// ISA layout: lanes0-15 M=lane, K chunks {k0..k0+7} (v0-3) and {k0+16..k0+23} (v4-7);
//             lanes16-31 M=lane-16, chunks {k0+8..15} and {k0+24..31}.
__device__ __forceinline__ v16bf load_a_tile(const unsigned short* sm, int ld,
                                             int m0, int k0, int lane) {
    const int half = lane >> 4, r = lane & 15;
    const unsigned short* row = sm + (m0 + r) * ld;
    union { v16bf v; u32x4 q[2]; } t;
    t.q[0] = *(const u32x4*)(row + k0 + half * 8);
    t.q[1] = *(const u32x4*)(row + k0 + 16 + half * 8);
    return t.v;
}

// B-matrix (32x16 bf16, KxN) tile where LDS source is stored N-major:
// src[n][k] == B[k][n], stride ld. ISA layout: lane n (0-15) holds column N=n,
// K chunks {k0..k0+7} and {k0+16..k0+23}; lanes16-31 hold {k0+8..15} and {k0+24..31}.
__device__ __forceinline__ v16bf load_b_tile(const unsigned short* sm, int ld,
                                             int n0, int k0, int lane) {
    const int half = lane >> 4, n = lane & 15;
    const unsigned short* row = sm + (n0 + n) * ld;
    union { v16bf v; u32x4 q[2]; } t;
    t.q[0] = *(const u32x4*)(row + k0 + half * 8);
    t.q[1] = *(const u32x4*)(row + k0 + 16 + half * 8);
    return t.v;
}

__device__ __forceinline__ v8f wmma_bf16(v16bf a, v16bf b, v8f c) {
    // (neg_a, A, neg_b, B, c_mod, C, reuse_a, reuse_b)
    return __builtin_amdgcn_wmma_f32_16x16x32_bf16(false, a, false, b, (short)0, c,
                                                   false, false);
}

// ---------- kernel 1: GroupNorm statistics ----------
// grid = B*G blocks of 256; group data is contiguous: 128*4096 f32 per (b,g)
__global__ void gn_stats_kernel(const float* __restrict__ x, float* __restrict__ stats) {
    const int bg = blockIdx.x;
    const f32x4* p4 = (const f32x4*)(x + (size_t)bg * (CG_ * HW_));
    float s = 0.f, ss = 0.f;
    for (int i = threadIdx.x; i < (CG_ * HW_) / 4; i += 256) {
        f32x4 v = p4[i];
        s  += v.x + v.y + v.z + v.w;
        ss += v.x * v.x + v.y * v.y + v.z * v.z + v.w * v.w;
    }
    __shared__ float red0[256], red1[256];
    const int tid = threadIdx.x;
    red0[tid] = s; red1[tid] = ss;
    __syncthreads();
    for (int off = 128; off > 0; off >>= 1) {
        if (tid < off) { red0[tid] += red0[tid + off]; red1[tid] += red1[tid + off]; }
        __syncthreads();
    }
    if (tid == 0) {
        const float n = (float)(CG_ * HW_);
        float mu  = red0[0] / n;
        float var = red1[0] / n - mu * mu;
        stats[bg * 2 + 0] = mu;
        stats[bg * 2 + 1] = rsqrtf(var + 1e-5f);
    }
}

// ---------- kernel 2: normalize + grouped qkv 1x1 conv (GEMM 384x128x4096 per b,g) ----------
// grid: x = pixel tile (64), y = o tile (6), z = b*G+g (64); block 128 (4 waves)
__global__ void qkv_kernel(const float* __restrict__ x, const float* __restrict__ stats,
                           const float* __restrict__ nw, const float* __restrict__ nb,
                           const float* __restrict__ qw, const float* __restrict__ qb,
                           unsigned short* __restrict__ qkv) {
    const int p0 = blockIdx.x * 64, o0 = blockIdx.y * 64;
    const int bg = blockIdx.z, b = bg >> 2, g = bg & 3;
    const int tid = threadIdx.x;

    __shared__ __align__(16) unsigned short As[64 * 136];  // [o][i]  (M x K)
    __shared__ __align__(16) unsigned short Bt[64 * 136];  // [p][i]  (N x K, N-major)

    // stage A: qkv_w[g][o0+o][i] -> bf16
    {
        const f32x4* w4 = (const f32x4*)(qw + ((size_t)g * 384 + o0) * CG_);
        for (int j = tid; j < 64 * CG_ / 4; j += 128) {
            f32x4 v = w4[j];
            const int e = j << 2;
            unsigned short* d = As + (e >> 7) * 136 + (e & 127);
            d[0] = f32_to_bf16(v.x); d[1] = f32_to_bf16(v.y);
            d[2] = f32_to_bf16(v.z); d[3] = f32_to_bf16(v.w);
        }
    }
    // stage B: xn[b][g*128+i][p0..p0+63] normalized, stored transposed Bt[p][i]
    {
        const float mu = stats[bg * 2], rstd = stats[bg * 2 + 1];
        const int ip = tid >> 1, ph = tid & 1;
        #pragma unroll
        for (int q = 0; q < 2; ++q) {
            const int i = ip * 2 + q, c = g * CG_ + i;
            const float a = nw[c] * rstd;
            const float b2 = nb[c] - mu * a;
            const f32x4* s4 = (const f32x4*)(x + ((size_t)b * C_ + c) * HW_ + p0 + ph * 32);
            #pragma unroll
            for (int j = 0; j < 8; ++j) {
                f32x4 v = s4[j];
                const int p = ph * 32 + j * 4;
                Bt[(p + 0) * 136 + i] = f32_to_bf16(v.x * a + b2);
                Bt[(p + 1) * 136 + i] = f32_to_bf16(v.y * a + b2);
                Bt[(p + 2) * 136 + i] = f32_to_bf16(v.z * a + b2);
                Bt[(p + 3) * 136 + i] = f32_to_bf16(v.w * a + b2);
            }
        }
    }
    __syncthreads();

    const int lane = tid & 31, wid = tid >> 5, m0 = wid * 16;
    const int half = lane >> 4, n = lane & 15;
    v8f acc[4] = {};
    for (int kt = 0; kt < CG_; kt += 32) {
        v16bf a = load_a_tile(As, 136, m0, kt, lane);
        #pragma unroll
        for (int nt = 0; nt < 4; ++nt) {
            v16bf bm = load_b_tile(Bt, 136, nt * 16, kt, lane);
            acc[nt] = wmma_bf16(a, bm, acc[nt]);
        }
    }
    #pragma unroll
    for (int v = 0; v < 8; ++v) {
        const int m = m0 + v + half * 8;
        const int ch = g * 384 + o0 + m;                   // reshaped channel index
        const float bias = qb[ch];
        unsigned short* dst = qkv + ((size_t)b * QKVC + ch) * HW_ + p0;
        #pragma unroll
        for (int nt = 0; nt < 4; ++nt)
            dst[nt * 16 + n] = f32_to_bf16(acc[nt][v] + bias);
    }
}

// ---------- kernel 3: per-channel attention over H (64x64x64 matmuls + softmax) ----------
// grid = B*C (8192) blocks of 128 (4 waves). Output overwrites the Q slot (bf16).
__global__ void attn_kernel(unsigned short* __restrict__ qkv) {
    const int bx = blockIdx.x, b = bx >> 9, c = bx & 511;
    const size_t base = (size_t)b * QKVC * HW_;
    unsigned short*       qp = qkv + base + (size_t)c * HW_;
    const unsigned short* kp = qkv + base + (size_t)(C_ + c) * HW_;
    const unsigned short* vp = qkv + base + (size_t)(2 * C_ + c) * HW_;

    __shared__ __align__(16) unsigned short Qs[64 * 72];   // [h][w]
    __shared__ __align__(16) unsigned short Ks[64 * 72];   // [h][w] (cols of K^T)
    __shared__ __align__(16) unsigned short Vt[64 * 72];   // [w][h] (cols of V)
    __shared__ __align__(16) unsigned short Ps[64 * 72];   // softmax(S) bf16
    __shared__ float Ss[64 * 65];

    const int tid = threadIdx.x;
    const u32x4* q4 = (const u32x4*)qp;
    const u32x4* k4 = (const u32x4*)kp;
    const u32x4* v4 = (const u32x4*)vp;
    for (int j = tid; j < 512; j += 128) {     // 512 * 8 bf16 = 4096 elems
        const int row = j >> 3, col = (j & 7) * 8;
        *(u32x4*)(Qs + row * 72 + col) = q4[j];
        *(u32x4*)(Ks + row * 72 + col) = k4[j];
        union { u32x4 q; unsigned short s[8]; } t; t.q = v4[j];
        #pragma unroll
        for (int e = 0; e < 8; ++e) Vt[(col + e) * 72 + row] = t.s[e];
    }
    __syncthreads();

    const int lane = tid & 31, wid = tid >> 5, m0 = wid * 16;
    const int half = lane >> 4, n = lane & 15;

    // S = Q K^T * hd^-0.5   (each wave: 16 rows x 64 cols)
    {
        v8f acc[4] = {};
        for (int kt = 0; kt < 64; kt += 32) {
            v16bf a = load_a_tile(Qs, 72, m0, kt, lane);
            #pragma unroll
            for (int nt = 0; nt < 4; ++nt) {
                v16bf bm = load_b_tile(Ks, 72, nt * 16, kt, lane);
                acc[nt] = wmma_bf16(a, bm, acc[nt]);
            }
        }
        #pragma unroll
        for (int v = 0; v < 8; ++v) {
            float* srow = Ss + (m0 + v + half * 8) * 65;
            #pragma unroll
            for (int nt = 0; nt < 4; ++nt)
                srow[nt * 16 + n] = acc[nt][v] * 0.125f;   // 64^-0.5
        }
    }
    __syncthreads();

    // softmax over y (row direction), one thread per row
    if (tid < 64) {
        float* row = Ss + tid * 65;
        float mx = row[0];
        for (int j = 1; j < 64; ++j) mx = fmaxf(mx, row[j]);
        float sum = 0.f;
        for (int j = 0; j < 64; ++j) { float e = __expf(row[j] - mx); row[j] = e; sum += e; }
        const float inv = 1.f / sum;
        unsigned short* pr = Ps + tid * 72;
        for (int j = 0; j < 64; ++j) pr[j] = f32_to_bf16(row[j] * inv);
    }
    __syncthreads();

    // O = P @ V  (B-tiles come from Vt: src[w][y] == V[y][w])
    {
        v8f acc[4] = {};
        for (int kt = 0; kt < 64; kt += 32) {
            v16bf a = load_a_tile(Ps, 72, m0, kt, lane);
            #pragma unroll
            for (int nt = 0; nt < 4; ++nt) {
                v16bf bm = load_b_tile(Vt, 72, nt * 16, kt, lane);
                acc[nt] = wmma_bf16(a, bm, acc[nt]);
            }
        }
        #pragma unroll
        for (int v = 0; v < 8; ++v) {
            unsigned short* dst = qp + (m0 + v + half * 8) * 64;
            #pragma unroll
            for (int nt = 0; nt < 4; ++nt)
                dst[nt * 16 + n] = f32_to_bf16(acc[nt][v]);
        }
    }
}

// ---------- kernel 4: grouped proj 1x1 conv (GEMM 128x128x4096 per b,g), out = 2*(conv+bias) ----------
// grid: x = pixel tile (64), y = o tile (2), z = b*G+g (64); block 128
__global__ void proj_kernel(const unsigned short* __restrict__ att,
                            const float* __restrict__ pw, const float* __restrict__ pb,
                            float* __restrict__ out) {
    const int p0 = blockIdx.x * 64, o0 = blockIdx.y * 64;
    const int bg = blockIdx.z, b = bg >> 2, g = bg & 3;
    const int tid = threadIdx.x;

    __shared__ __align__(16) unsigned short As[64 * 136];  // [o][i]
    __shared__ __align__(16) unsigned short Bt[64 * 136];  // [p][i]

    {
        const f32x4* w4 = (const f32x4*)(pw + ((size_t)g * CG_ + o0) * CG_);
        for (int j = tid; j < 64 * CG_ / 4; j += 128) {
            f32x4 v = w4[j];
            const int e = j << 2;
            unsigned short* d = As + (e >> 7) * 136 + (e & 127);
            d[0] = f32_to_bf16(v.x); d[1] = f32_to_bf16(v.y);
            d[2] = f32_to_bf16(v.z); d[3] = f32_to_bf16(v.w);
        }
    }
    {
        const int ip = tid >> 1, ph = tid & 1;
        #pragma unroll
        for (int q = 0; q < 2; ++q) {
            const int i = ip * 2 + q;
            const u32x4* s4 = (const u32x4*)(att + ((size_t)b * QKVC + g * CG_ + i) * HW_
                                             + p0 + ph * 32);
            #pragma unroll
            for (int j = 0; j < 4; ++j) {                  // 4 * 8 bf16 = 32 pixels
                union { u32x4 q; unsigned short s[8]; } t; t.q = s4[j];
                const int p = ph * 32 + j * 8;
                #pragma unroll
                for (int e = 0; e < 8; ++e) Bt[(p + e) * 136 + i] = t.s[e];
            }
        }
    }
    __syncthreads();

    const int lane = tid & 31, wid = tid >> 5, m0 = wid * 16;
    const int half = lane >> 4, n = lane & 15;
    v8f acc[4] = {};
    for (int kt = 0; kt < CG_; kt += 32) {
        v16bf a = load_a_tile(As, 136, m0, kt, lane);
        #pragma unroll
        for (int nt = 0; nt < 4; ++nt) {
            v16bf bm = load_b_tile(Bt, 136, nt * 16, kt, lane);
            acc[nt] = wmma_bf16(a, bm, acc[nt]);
        }
    }
    #pragma unroll
    for (int v = 0; v < 8; ++v) {
        const int m = m0 + v + half * 8;
        const int ch = g * CG_ + o0 + m;
        const float bias = pb[ch];
        float* dst = out + ((size_t)b * C_ + ch) * HW_ + p0;
        #pragma unroll
        for (int nt = 0; nt < 4; ++nt)
            dst[nt * 16 + n] = 2.f * (acc[nt][v] + bias);  // module returns out + out
    }
}

// ---------- launch ----------
extern "C" void kernel_launch(void* const* d_in, const int* in_sizes, int n_in,
                              void* d_out, int out_size, void* d_ws, size_t ws_size,
                              hipStream_t stream) {
    (void)in_sizes; (void)n_in; (void)out_size; (void)ws_size;
    const float* x      = (const float*)d_in[0];
    const float* norm_w = (const float*)d_in[1];
    const float* norm_b = (const float*)d_in[2];
    const float* qkv_w  = (const float*)d_in[3];
    const float* qkv_b  = (const float*)d_in[4];
    const float* proj_w = (const float*)d_in[5];
    const float* proj_b = (const float*)d_in[6];
    float* out = (float*)d_out;

    float* stats = (float*)d_ws;                                   // 64*2 f32
    unsigned short* qkvbuf = (unsigned short*)((char*)d_ws + 512); // B*1536*4096 bf16 (~192MB)

    gn_stats_kernel<<<B_ * G_, 256, 0, stream>>>(x, stats);
    qkv_kernel<<<dim3(HW_ / 64, 6, B_ * G_), 128, 0, stream>>>(
        x, stats, norm_w, norm_b, qkv_w, qkv_b, qkvbuf);
    attn_kernel<<<B_ * C_, 128, 0, stream>>>(qkvbuf);
    proj_kernel<<<dim3(HW_ / 64, 2, B_ * G_), 128, 0, stream>>>(
        qkvbuf, proj_w, proj_b, out);
}
